// self_loop_attention_GCN_65463891526375
// MI455X (gfx1250) — compile-verified, matrix operand order checked by reading, not run on hardware
//
#include <hip/hip_runtime.h>
#include <hip/hip_bf16.h>

#define NROW   10000
#define NF     256
#define KP     10112          // 79 * 128, zero-padded K for the big GEMM
#define KC     128            // K chunk staged in LDS per iteration
#define LDSROW 136            // 128 + 8 halves pad -> conflict-free ds_load_b128
#define MBLK   64             // rows per block in main GEMM
#define NBLKS  157            // ceil(10000 / 64)

typedef __attribute__((ext_vector_type(16))) _Float16 v16h;
typedef __attribute__((ext_vector_type(8)))  _Float16 v8h;
typedef __attribute__((ext_vector_type(4)))  _Float16 v4h;
typedef __attribute__((ext_vector_type(8)))  float    v8f;
typedef __attribute__((ext_vector_type(4)))  float    v4f;

// ---------------------------------------------------------------------------
// Kernel 1: deg = row-sum(adj); dinv = deg > 0 ? deg^-0.5 : 0
// ---------------------------------------------------------------------------
__global__ __launch_bounds__(256)
void k_rowdeg(const float* __restrict__ adj, float* __restrict__ dinv)
{
    const int r = blockIdx.x;
    const float* rowp = adj + (long)r * NROW;
    float s = 0.f;
    for (int i = threadIdx.x; i < NROW / 4; i += 256) {
        v4f v = *(const v4f*)(rowp + i * 4);
        s += v.x + v.y + v.z + v.w;
    }
    for (int off = 16; off > 0; off >>= 1) s += __shfl_down(s, off, 32);
    __shared__ float red[8];
    const int lane = threadIdx.x & 31, w = threadIdx.x >> 5;
    if (lane == 0) red[w] = s;
    __syncthreads();
    if (threadIdx.x == 0) {
        float t = 0.f;
        for (int i = 0; i < 8; i++) t += red[i];
        dinv[r] = t > 0.f ? rsqrtf(t) : 0.f;
    }
}

// ---------------------------------------------------------------------------
// Kernel 2: Wt[c][k] = (f16) W[k][c]   (tiled LDS transpose, 256x256)
// ---------------------------------------------------------------------------
__global__ __launch_bounds__(256)
void k_transW(const float* __restrict__ W, _Float16* __restrict__ Wt)
{
    __shared__ float t[32][33];
    const int bx = blockIdx.x & 7, by = blockIdx.x >> 3;      // 8x8 tiles
    const int x = threadIdx.x & 31, y = threadIdx.x >> 5;     // 32x8 threads
    for (int i = 0; i < 32; i += 8)
        t[y + i][x] = W[(by * 32 + y + i) * NF + bx * 32 + x];
    __syncthreads();
    for (int i = 0; i < 32; i += 8)
        Wt[(bx * 32 + y + i) * NF + by * 32 + x] = (_Float16)t[x][y + i];
}

// ---------------------------------------------------------------------------
// Kernel 3: supT[c][m] = (f16)( dinv[m] * (features @ W)[m][c] ), m padded->KP
// 16 waves/block, wave w owns output columns [w*16, w*16+16), 16 rows/block.
// ---------------------------------------------------------------------------
__global__ __launch_bounds__(512)
void k_support(const float* __restrict__ feat, const _Float16* __restrict__ Wt,
               const float* __restrict__ dinv, _Float16* __restrict__ supT)
{
    const int rb  = blockIdx.x;
    const int tid = threadIdx.x;

    if (rb >= NROW / 16) {                 // zero the K padding rows
        #pragma unroll
        for (int j = 0; j < 8; j++) {
            int lin = tid * 8 + j;         // 0..4095 over 256 cols x 16 rows
            int c = lin >> 4, mi = lin & 15;
            supT[(long)c * KP + rb * 16 + mi] = (_Float16)0.f;
        }
        return;
    }

    __shared__ _Float16 Fs[16 * 264];      // 16 rows x (256+8 pad) halves

    const int lane = tid & 31;
    const int wave = tid >> 5;
    const int r16  = lane & 15;
    const int hi   = (lane >> 4) & 1;

    // stage features tile [16][256] -> f16 LDS
    {
        const int srow = tid >> 5;
        const int sc   = (tid & 31) * 8;
        const float* fp = feat + (long)(rb * 16 + srow) * NF + sc;
        v4f f0 = *(const v4f*)(fp);
        v4f f1 = *(const v4f*)(fp + 4);
        v8h h;
        h[0]=(_Float16)f0.x; h[1]=(_Float16)f0.y; h[2]=(_Float16)f0.z; h[3]=(_Float16)f0.w;
        h[4]=(_Float16)f1.x; h[5]=(_Float16)f1.y; h[6]=(_Float16)f1.z; h[7]=(_Float16)f1.w;
        *(v8h*)(&Fs[srow * 264 + sc]) = h;
    }
    __syncthreads();

    const int col = wave * 16 + r16;
    const _Float16* bcol = Wt + (long)col * NF;

    v8f acc = {};
    #pragma unroll
    for (int kk = 0; kk < NF; kk += 32) {
        const _Float16* ap = &Fs[r16 * 264 + kk + hi * 8];
        v8h a0 = *(const v8h*)(ap);
        v8h a1 = *(const v8h*)(ap + 16);
        const _Float16* bp = bcol + kk + hi * 8;
        v8h b0 = *(const v8h*)(bp);
        v8h b1 = *(const v8h*)(bp + 16);
        v16h a, b;
        #pragma unroll
        for (int i = 0; i < 8; i++) { a[i]=a0[i]; a[i+8]=a1[i]; b[i]=b0[i]; b[i+8]=b1[i]; }
        acc = __builtin_amdgcn_wmma_f32_16x16x32_f16(false, a, false, b,
                                                     (short)0, acc, false, false);
    }

    // scaled, packed f16 store: 8 consecutive m per half-wave
    const int mb = rb * 16 + hi * 8;
    v8h o;
    #pragma unroll
    for (int v = 0; v < 8; v++) o[v] = (_Float16)(dinv[mb + v] * acc[v]);
    *(v8h*)(supT + (long)col * KP + mb) = o;
}

// ---------------------------------------------------------------------------
// Kernel 4: out = relu( dinv[r]*(f16(adj[r,:]) @ supT  + 2*supT[r,:]) + b )
// 157 blocks x 512 threads (16 waves). Block tile: 64 rows x 256 cols.
// Wave grid 2(M) x 8(N): each wave owns a 32x32 output tile -> 4 independent
// WMMA accumulator chains per k-step, A reused across 2 column tiles,
// B reused across 2 row tiles. Double-buffered 64x128 f16 A in LDS.
// ---------------------------------------------------------------------------
__global__ __launch_bounds__(512)
void k_gcn_main(const float* __restrict__ adj, const _Float16* __restrict__ supT,
                const float* __restrict__ dinv, const float* __restrict__ bias,
                float* __restrict__ out)
{
    __shared__ _Float16 As[2][MBLK * LDSROW];

    const int tid  = threadIdx.x;
    const int lane = tid & 31;
    const int wave = tid >> 5;
    const int rb   = blockIdx.x;
    const int r16  = lane & 15;
    const int hi   = (lane >> 4) & 1;

    const int mgrp = wave >> 3;            // 0..1 : row-tile pair [mgrp*32, +32)
    const int ngrp = wave & 7;             // 0..7 : col group [ngrp*32, +32)

    // staging map: thread -> (row 0..63, 16 consecutive K columns)
    const int  srow = tid >> 3;            // 0..63
    const int  scol = (tid & 7) * 16;      // 0,16,...,112
    const int  arow = rb * MBLK + srow;
    const long arowbase = (long)arow * NROW;
    const bool arow_ok  = arow < NROW;

    const int col0 = ngrp * 32 + r16;
    const int col1 = col0 + 16;
    const _Float16* bcol0 = supT + (long)col0 * KP;
    const _Float16* bcol1 = supT + (long)col1 * KP;

    v8f acc00 = {}, acc01 = {}, acc10 = {}, acc11 = {};

    auto cvt8 = [](v4f a, v4f b) {
        v8h h;
        h[0]=(_Float16)a.x; h[1]=(_Float16)a.y; h[2]=(_Float16)a.z; h[3]=(_Float16)a.w;
        h[4]=(_Float16)b.x; h[5]=(_Float16)b.y; h[6]=(_Float16)b.z; h[7]=(_Float16)b.w;
        return h;
    };

    auto stage = [&](int buf, int k0) {
        const int k = k0 + scol;                     // multiple of 16
        v4f f0 = {}, f1 = {}, f2 = {}, f3 = {};
        if (arow_ok && k < NROW) {                   // 10000 % 16 == 0
            const float* p = adj + arowbase + k;
            f0 = *(const v4f*)(p);
            f1 = *(const v4f*)(p + 4);
            f2 = *(const v4f*)(p + 8);
            f3 = *(const v4f*)(p + 12);
        }
        _Float16* d = &As[buf][srow * LDSROW + scol];
        *(v8h*)(d)     = cvt8(f0, f1);
        *(v8h*)(d + 8) = cvt8(f2, f3);
    };

    auto dostep = [&](int buf, int kk, int kg) {
        const _Float16* a0p = &As[buf][(mgrp * 32 + r16) * LDSROW + kk + hi * 8];
        const _Float16* a1p = a0p + 16 * LDSROW;
        v8h a0l = *(const v8h*)(a0p); v8h a0h = *(const v8h*)(a0p + 16);
        v8h a1l = *(const v8h*)(a1p); v8h a1h = *(const v8h*)(a1p + 16);
        const _Float16* b0p = bcol0 + kg + hi * 8;
        const _Float16* b1p = bcol1 + kg + hi * 8;
        v8h b0l = *(const v8h*)(b0p); v8h b0h = *(const v8h*)(b0p + 16);
        v8h b1l = *(const v8h*)(b1p); v8h b1h = *(const v8h*)(b1p + 16);
        v16h A0, A1, B0, B1;
        #pragma unroll
        for (int i = 0; i < 8; i++) {
            A0[i]=a0l[i]; A0[i+8]=a0h[i];  A1[i]=a1l[i]; A1[i+8]=a1h[i];
            B0[i]=b0l[i]; B0[i+8]=b0h[i];  B1[i]=b1l[i]; B1[i+8]=b1h[i];
        }
        acc00 = __builtin_amdgcn_wmma_f32_16x16x32_f16(false, A0, false, B0, (short)0, acc00, false, false);
        acc01 = __builtin_amdgcn_wmma_f32_16x16x32_f16(false, A0, false, B1, (short)0, acc01, false, false);
        acc10 = __builtin_amdgcn_wmma_f32_16x16x32_f16(false, A1, false, B0, (short)0, acc10, false, false);
        acc11 = __builtin_amdgcn_wmma_f32_16x16x32_f16(false, A1, false, B1, (short)0, acc11, false, false);
    };

    stage(0, 0);
    __syncthreads();

    const int nCh = KP / KC;               // 79
    for (int ch = 0; ch < nCh; ++ch) {
        const int p  = ch & 1;
        if (ch + 1 < nCh) stage(p ^ 1, (ch + 1) * KC);
        const int kg = ch * KC;
        dostep(p, 0,  kg + 0);
        dostep(p, 32, kg + 32);
        dostep(p, 64, kg + 64);
        dostep(p, 96, kg + 96);
        __syncthreads();
    }

    // epilogue: diagonal (att+1 == 2) correction, dinv[r] scale, bias, ReLU
    auto epil = [&](const v8f& a, int mt, int c, float bc) {
        const int rowbase = rb * MBLK + mgrp * 32 + mt * 16 + hi * 8;
        #pragma unroll
        for (int v = 0; v < 8; v++) {
            const int row = rowbase + v;
            if (row < NROW) {
                const float dv   = dinv[row];
                const float corr = 2.0f * (float)supT[(long)c * KP + row];
                float val = dv * (a[v] + corr) + bc;
                out[(long)row * NF + c] = val > 0.f ? val : 0.f;
            }
        }
    };
    const float bc0 = bias[col0], bc1 = bias[col1];
    epil(acc00, 0, col0, bc0);
    epil(acc01, 0, col1, bc1);
    epil(acc10, 1, col0, bc0);
    epil(acc11, 1, col1, bc1);
}

// ---------------------------------------------------------------------------
extern "C" void kernel_launch(void* const* d_in, const int* in_sizes, int n_in,
                              void* d_out, int out_size, void* d_ws, size_t ws_size,
                              hipStream_t stream)
{
    (void)in_sizes; (void)n_in; (void)out_size; (void)ws_size;

    const float* adj  = (const float*)d_in[0];   // [N,N]
    const float* feat = (const float*)d_in[1];   // [N,256]
    const float* W    = (const float*)d_in[2];   // [256,256]
    const float* b    = (const float*)d_in[3];   // [256]
    // d_in[4] (a) only feeds a size-1 softmax == 1.0; mathematically unused.

    char* ws = (char*)d_ws;
    float*    dinv = (float*)ws;                               // 40,000 B
    _Float16* supT = (_Float16*)(ws + 40000);                  // 256*KP*2 = 5,177,344 B
    _Float16* Wt   = (_Float16*)(ws + 40000 + (size_t)256 * KP * 2); // 131,072 B
    float*    out  = (float*)d_out;

    k_rowdeg  <<<NROW,    256, 0, stream>>>(adj, dinv);
    k_transW  <<<64,      256, 0, stream>>>(W, Wt);
    k_support <<<KP / 16, 512, 0, stream>>>(feat, Wt, dinv, supT);
    k_gcn_main<<<NBLKS,   512, 0, stream>>>(adj, supT, dinv, b, out);
}
